// MultiQueryAttention_20822001451538
// MI455X (gfx1250) — matrix operand order repaired and checked by reading, not compiled, not run on hardware
//
#include <hip/hip_runtime.h>
#include <hip/hip_bf16.h>
#include <stdint.h>

// ---------------------------------------------------------------------------
// MQA forward for MI455X (gfx1250, wave32, WMMA).
// All heavy matmuls use v_wmma_f32_16x16x32_bf16 (bf16 in, fp32 accumulate).
// ---------------------------------------------------------------------------

typedef __bf16 bf16_t;
typedef __attribute__((ext_vector_type(16))) __bf16 v16bf;
typedef __attribute__((ext_vector_type(8)))  __bf16 v8bf;
typedef __attribute__((ext_vector_type(8)))  float  v8f;

static constexpr int BB    = 4;
static constexpr int TT    = 2048;
static constexpr int NH    = 16;
static constexpr int DH    = 128;
static constexpr int MODEL = 2048;
static constexpr int BT    = BB * TT;

__device__ __forceinline__ bf16_t f2bf(float f) {
  union { float f; uint32_t u; } cv; cv.f = f;
  uint32_t u = cv.u;
  u += 0x7FFFu + ((u >> 16) & 1u);           // round-to-nearest-even
  union { uint16_t s; bf16_t b; } o; o.s = (uint16_t)(u >> 16);
  return o.b;
}

__global__ void cvt_f32_to_bf16(const float* __restrict__ in,
                                bf16_t* __restrict__ out, int n) {
  int i = blockIdx.x * blockDim.x + threadIdx.x;
  if (i < n) out[i] = f2bf(in[i]);
}

// ---------------------------------------------------------------------------
// GEMM: out[M x 2048] = (A[M x 2048](bf16) * B[2048 x 2048](bf16) + bias)*scl
// Block tile 128x128, 8 waves of 32x64 each (2x4 grid of 16x16 WMMA tiles).
// K staged 64 at a time (two WMMA K-steps per barrier pair).
// ---------------------------------------------------------------------------
template <bool OUT_BF16>
__global__ __launch_bounds__(256)
void gemm_bf16_wmma(const bf16_t* __restrict__ A,    // [M x K] row-major
                    const bf16_t* __restrict__ Bm,   // [K x N] row-major
                    const float*  __restrict__ bias, // [N]
                    void* __restrict__ outp,
                    float outScale) {
  constexpr int N = 2048, K = 2048;
  constexpr int LDA = 72, LDB = 72;   // padded LDS strides (elements, 144B rows)
  __shared__ alignas(16) bf16_t As[128 * LDA];      // As[r][k], k in 0..63
  __shared__ alignas(16) bf16_t Bs[128 * LDB];      // transposed: Bs[n][k]

  const int tid  = threadIdx.x;
  const int lane = tid & 31;
  const int wave = tid >> 5;     // 0..7
  const int wr   = wave >> 1;    // 0..3 : row-wave (32 rows each)
  const int wc   = wave & 1;     // 0..1 : col-wave (64 cols each)
  const int m0   = blockIdx.y * 128;
  const int n0   = blockIdx.x * 128;

  const int l16 = lane & 15;
  const int lhi = lane >> 4;     // 0 or 1
  const int klo = lhi * 8;       // A-frag K base
  const int k16 = lhi * 16;      // unused (B frags use klo)

  v8f acc[2][4];
  #pragma unroll
  for (int mt = 0; mt < 2; ++mt)
    #pragma unroll
    for (int nt = 0; nt < 4; ++nt)
      #pragma unroll
      for (int j = 0; j < 8; ++j) acc[mt][nt][j] = 0.0f;

  const int ar = tid >> 1;          // 0..127  (A staging row)
  const int ac = (tid & 1) * 32;    // 0 / 32  (A staging col chunk)
  const int bk = tid & 63;          // 0..63   (B staging k)
  const int bn = (tid >> 6) * 32;   // 0..96   (B staging n chunk)

  for (int kb = 0; kb < K; kb += 64) {
    __syncthreads();
    // stage A block [128 x 64]
    {
      const v8bf* gA = (const v8bf*)(A + (size_t)(m0 + ar) * K + kb + ac);
      #pragma unroll
      for (int c = 0; c < 4; ++c)
        *(v8bf*)&As[ar * LDA + ac + c * 8] = gA[c];
    }
    // stage B block [64 x 128], transposed into Bs[n][k]
    {
      const bf16_t* gB = Bm + (size_t)(kb + bk) * N + n0 + bn;
      #pragma unroll
      for (int c = 0; c < 4; ++c) {
        v8bf bv = ((const v8bf*)gB)[c];
        #pragma unroll
        for (int j = 0; j < 8; ++j)
          Bs[(bn + c * 8 + j) * LDB + bk] = bv[j];
      }
    }
    if (kb + 64 < K) {
      __builtin_prefetch(A  + (size_t)(m0 + ar) * K + kb + 64 + ac, 0, 1);
      __builtin_prefetch(Bm + (size_t)(kb + 64 + bk) * N + n0 + bn, 0, 1);
    }
    __syncthreads();

    #pragma unroll
    for (int ks = 0; ks < 64; ks += 32) {
      // A fragments (lane m = lane%16, K = klo+{0..7}, klo+16+{0..7})
      v16bf afr[2];
      #pragma unroll
      for (int mt = 0; mt < 2; ++mt) {
        const int row = wr * 32 + mt * 16 + l16;
        v8bf lo = *(const v8bf*)&As[row * LDA + ks + klo];
        v8bf hi = *(const v8bf*)&As[row * LDA + ks + klo + 16];
        #pragma unroll
        for (int j = 0; j < 8; ++j) { afr[mt][j] = lo[j]; afr[mt][8 + j] = hi[j]; }
      }
      // B fragments (lane n = lane%16, K = klo+{0..7}, klo+16+{0..7})
      v16bf bfr[4];
      #pragma unroll
      for (int nt = 0; nt < 4; ++nt) {
        const int col = wc * 64 + nt * 16 + l16;
        v8bf lo = *(const v8bf*)&Bs[col * LDB + ks + klo];
        v8bf hi = *(const v8bf*)&Bs[col * LDB + ks + klo + 16];
        #pragma unroll
        for (int j = 0; j < 8; ++j) { bfr[nt][j] = lo[j]; bfr[nt][8 + j] = hi[j]; }
      }
      #pragma unroll
      for (int mt = 0; mt < 2; ++mt)
        #pragma unroll
        for (int nt = 0; nt < 4; ++nt)
          acc[mt][nt] = __builtin_amdgcn_wmma_f32_16x16x32_bf16(
              false, afr[mt], false, bfr[nt], (short)0, acc[mt][nt], false, false);
    }
  }

  // epilogue: C/D layout M = r + 8*(lane>=16), N = lane%16
  #pragma unroll
  for (int mt = 0; mt < 2; ++mt)
    #pragma unroll
    for (int nt = 0; nt < 4; ++nt) {
      const int gCol = n0 + wc * 64 + nt * 16 + l16;
      const float bv = bias[gCol];
      #pragma unroll
      for (int r = 0; r < 8; ++r) {
        const int gRow = m0 + wr * 32 + mt * 16 + r + lhi * 8;
        const float v = (acc[mt][nt][r] + bv) * outScale;
        if constexpr (OUT_BF16)
          ((bf16_t*)outp)[(size_t)gRow * N + gCol] = f2bf(v);
        else
          ((float*)outp)[(size_t)gRow * N + gCol] = v;
      }
    }
}

// ---------------------------------------------------------------------------
// Flash-style MQA attention. One block per (b, h, 128 q-rows); each wave owns
// 16 q-rows. 64 keys per iteration staged in LDS (V transposed) to amortize
// the softmax shuffle/exp work over 32 WMMAs per iteration.
// NOTE: q is pre-scaled by 1/sqrt(d) in the Q-projection GEMM epilogue.
// ---------------------------------------------------------------------------
__global__ __launch_bounds__(256)
void mqa_attention_wmma(const bf16_t* __restrict__ Q,   // [B*T, 2048] = (b,t,h,d)
                        const bf16_t* __restrict__ Kb,  // [B*T, 128]
                        const bf16_t* __restrict__ Vb,  // [B*T, 128]
                        bf16_t* __restrict__ Ob) {      // [B*T, 2048]
  constexpr int KBLK = 64;  // keys per iteration
  constexpr int LDK  = 136; // Ks[key][d]   stride
  constexpr int LDV  = 72;  // Vt[d][key]   stride (64 keys + pad)
  constexpr int LDP  = 72;  // Ps[row][key] stride
  __shared__ alignas(16) bf16_t Ks[KBLK * LDK];     // 17408 B
  __shared__ alignas(16) bf16_t Vt[128 * LDV];      // 18432 B
  __shared__ alignas(16) bf16_t Ps[8][16 * LDP];    // 18432 B

  const int tid  = threadIdx.x;
  const int lane = tid & 31;
  const int wave = tid >> 5;
  const int b    = blockIdx.z;
  const int h    = blockIdx.y;
  const int q0   = blockIdx.x * 128;
  const int l16  = lane & 15;
  const int lhi  = lane >> 4;
  const int klo  = lhi * 8;
  const int k16  = lhi * 16;

  // load Q fragments for this wave's 16 rows (d = 128 -> 4 K-steps of 32)
  v16bf qf[4];
  {
    const int t = q0 + wave * 16 + l16;
    const bf16_t* qrow = Q + (size_t)(b * TT + t) * MODEL + h * DH;
    #pragma unroll
    for (int kk = 0; kk < 4; ++kk) {
      v8bf lo = *(const v8bf*)(qrow + kk * 32 + klo);
      v8bf hi = *(const v8bf*)(qrow + kk * 32 + klo + 16);
      #pragma unroll
      for (int j = 0; j < 8; ++j) { qf[kk][j] = lo[j]; qf[kk][8 + j] = hi[j]; }
    }
  }

  v8f o[8];
  #pragma unroll
  for (int nt = 0; nt < 8; ++nt)
    #pragma unroll
    for (int j = 0; j < 8; ++j) o[nt][j] = 0.0f;
  float mrow[8], lrow[8];
  #pragma unroll
  for (int r = 0; r < 8; ++r) { mrow[r] = -INFINITY; lrow[r] = 0.0f; }

  const int skr = tid >> 2;          // K staging: key row 0..63
  const int skd = (tid & 3) * 32;    //            d chunk {0,32,64,96}
  const int svk = tid & 63;          // V staging: key row 0..63
  const int svd = (tid >> 6) * 32;   //            d chunk {0,32,64,96}

  for (int kb = 0; kb < TT; kb += KBLK) {
    __syncthreads();
    // stage K block as-is, V block transposed
    {
      const v8bf* gk = (const v8bf*)(Kb + (size_t)(b * TT + kb + skr) * DH + skd);
      #pragma unroll
      for (int c = 0; c < 4; ++c)
        *(v8bf*)&Ks[skr * LDK + skd + c * 8] = gk[c];
      const bf16_t* gv = Vb + (size_t)(b * TT + kb + svk) * DH + svd;
      #pragma unroll
      for (int c = 0; c < 4; ++c) {
        v8bf vv = ((const v8bf*)gv)[c];
        #pragma unroll
        for (int j = 0; j < 8; ++j)
          Vt[(svd + c * 8 + j) * LDV + svk] = vv[j];
      }
    }
    __syncthreads();

    // S (16 q-rows x 64 keys) = Q * K^T ; 4 n-tiles x 4 k-steps
    v8f s[4];
    #pragma unroll
    for (int nt = 0; nt < 4; ++nt) {
      v8f z;
      #pragma unroll
      for (int j = 0; j < 8; ++j) z[j] = 0.0f;
      const int krow = nt * 16 + l16;
      #pragma unroll
      for (int kk = 0; kk < 4; ++kk) {
        v8bf lo = *(const v8bf*)&Ks[krow * LDK + kk * 32 + klo];
        v8bf hi = *(const v8bf*)&Ks[krow * LDK + kk * 32 + klo + 16];
        v16bf kf;
        #pragma unroll
        for (int j = 0; j < 8; ++j) { kf[j] = lo[j]; kf[8 + j] = hi[j]; }
        z = __builtin_amdgcn_wmma_f32_16x16x32_bf16(
            false, qf[kk], false, kf, (short)0, z, false, false);
      }
      s[nt] = z;
    }

    // online softmax per row; rows 0-7 in lanes 0-15, rows 8-15 in lanes 16-31
    float corr[8];
    #pragma unroll
    for (int r = 0; r < 8; ++r) {
      float vmax = fmaxf(fmaxf(s[0][r], s[1][r]), fmaxf(s[2][r], s[3][r]));
      #pragma unroll
      for (int off = 8; off >= 1; off >>= 1)
        vmax = fmaxf(vmax, __shfl_xor(vmax, off, 32));
      const float mnew = fmaxf(mrow[r], vmax);
      corr[r] = __expf(mrow[r] - mnew);
      mrow[r] = mnew;
      const int M = r + lhi * 8;
      float rs = 0.0f;
      #pragma unroll
      for (int nt = 0; nt < 4; ++nt) {
        const float p = __expf(s[nt][r] - mnew);
        rs += p;
        Ps[wave][M * LDP + nt * 16 + l16] = f2bf(p);
      }
      #pragma unroll
      for (int off = 8; off >= 1; off >>= 1)
        rs += __shfl_xor(rs, off, 32);
      lrow[r] = lrow[r] * corr[r] + rs;
    }
    // rescale running O
    #pragma unroll
    for (int nt = 0; nt < 8; ++nt)
      #pragma unroll
      for (int r = 0; r < 8; ++r) o[nt][r] *= corr[r];

    __syncthreads();  // P (C-layout) -> LDS -> A-layout round trip

    // O += P * V ; two P fragments (K=32 each) x 8 d-tiles
    #pragma unroll
    for (int pk = 0; pk < 2; ++pk) {
      v16bf pf;
      {
        v8bf lo = *(const v8bf*)&Ps[wave][l16 * LDP + pk * 32 + klo];
        v8bf hi = *(const v8bf*)&Ps[wave][l16 * LDP + pk * 32 + klo + 16];
        #pragma unroll
        for (int j = 0; j < 8; ++j) { pf[j] = lo[j]; pf[8 + j] = hi[j]; }
      }
      #pragma unroll
      for (int nt = 0; nt < 8; ++nt) {
        const int dcol = nt * 16 + l16;
        v8bf lo = *(const v8bf*)&Vt[dcol * LDV + pk * 32 + klo];
        v8bf hi = *(const v8bf*)&Vt[dcol * LDV + pk * 32 + klo + 16];
        v16bf vf;
        #pragma unroll
        for (int j = 0; j < 8; ++j) { vf[j] = lo[j]; vf[8 + j] = hi[j]; }
        o[nt] = __builtin_amdgcn_wmma_f32_16x16x32_bf16(
            false, pf, false, vf, (short)0, o[nt], false, false);
      }
    }
  }

  // normalize and store: attn[b, t, h*128 + d] as bf16
  #pragma unroll
  for (int nt = 0; nt < 8; ++nt) {
    const int col = h * DH + nt * 16 + l16;
    #pragma unroll
    for (int r = 0; r < 8; ++r) {
      const int row = q0 + wave * 16 + r + lhi * 8;
      Ob[(size_t)(b * TT + row) * MODEL + col] = f2bf(o[nt][r] / lrow[r]);
    }
  }
}

// ---------------------------------------------------------------------------
extern "C" void kernel_launch(void* const* d_in, const int* in_sizes, int n_in,
                              void* d_out, int out_size, void* d_ws, size_t ws_size,
                              hipStream_t stream) {
  (void)in_sizes; (void)n_in; (void)out_size; (void)ws_size;
  const float* query = (const float*)d_in[0];
  const float* key   = (const float*)d_in[1];
  const float* value = (const float*)d_in[2];
  const float* Wq    = (const float*)d_in[3];
  const float* bq    = (const float*)d_in[4];
  const float* Wo    = (const float*)d_in[5];
  const float* bo    = (const float*)d_in[6];
  float* out = (float*)d_out;

  char* ws = (char*)d_ws;
  size_t off = 0;
  auto alloc = [&](size_t bytes) -> char* {
    char* p = ws + off;
    off += (bytes + 255) & ~(size_t)255;
    return p;
  };
  bf16_t* querybf = (bf16_t*)alloc((size_t)BT * MODEL * 2);
  bf16_t* Wqbf    = (bf16_t*)alloc((size_t)MODEL * MODEL * 2);
  bf16_t* Wobf    = (bf16_t*)alloc((size_t)MODEL * MODEL * 2);
  bf16_t* keybf   = (bf16_t*)alloc((size_t)BT * DH * 2);
  bf16_t* valbf   = (bf16_t*)alloc((size_t)BT * DH * 2);
  bf16_t* qbf     = (bf16_t*)alloc((size_t)BT * MODEL * 2);
  bf16_t* attnbf  = (bf16_t*)alloc((size_t)BT * MODEL * 2);

  auto cvt = [&](const float* src, bf16_t* dst, int n) {
    cvt_f32_to_bf16<<<dim3((n + 255) / 256), dim3(256), 0, stream>>>(src, dst, n);
  };
  cvt(query, querybf, BT * MODEL);
  cvt(Wq,    Wqbf,    MODEL * MODEL);
  cvt(Wo,    Wobf,    MODEL * MODEL);
  cvt(key,   keybf,   BT * DH);
  cvt(value, valbf,   BT * DH);

  // q = (query @ Wq + bq) * (1/sqrt(d))  -> bf16 [B*T, H*D]
  // (scale folded here so the attention hot loop does no scaling)
  gemm_bf16_wmma<true><<<dim3(MODEL / 128, BT / 128), 256, 0, stream>>>(
      querybf, Wqbf, bq, qbf, 0.08838834764831845f);
  // flash MQA attention -> bf16 [B*T, H*D]
  mqa_attention_wmma<<<dim3(TT / 128, NH, BB), 256, 0, stream>>>(
      qbf, keybf, valbf, attnbf);
  // out = attn @ Wo + bo -> fp32
  gemm_bf16_wmma<false><<<dim3(MODEL / 128, BT / 128), 256, 0, stream>>>(
      attnbf, Wobf, bo, out, 1.0f);
}
